// Model_81922206204437
// MI455X (gfx1250) — matrix operand order repaired and checked by reading, not compiled, not run on hardware
//
#include <hip/hip_runtime.h>

#define B_ 64
#define T_ 2048
#define H_ 128

typedef __attribute__((ext_vector_type(16))) __bf16 bf16x16;
typedef __attribute__((ext_vector_type(8)))  float  f32x8;

union Frag {
    bf16x16 v;
    uint4   q[2];
};

static __device__ __forceinline__ unsigned short f2bf(float f) {
    unsigned int u = __float_as_uint(f);
    unsigned int r = (u + 0x7FFFu + ((u >> 16) & 1u)) >> 16;
    return (unsigned short)r;
}
static __device__ __forceinline__ float bf2f(unsigned short h) {
    return __uint_as_float(((unsigned int)h) << 16);
}
static __device__ __forceinline__ float sigmoidf_(float x) {
    return 1.0f / (1.0f + __expf(-x));
}
static __device__ __forceinline__ float tanh_fast(float x) {
    // tanh(x) = 1 - 2/(exp(2x)+1); stays on v_exp_f32 fast path
    return 1.0f - 2.0f / (__expf(2.0f * x) + 1.0f);
}
static __device__ __forceinline__ unsigned lds_addr32(const void* p) {
    // generic (flat) LDS pointer: low 32 bits are the LDS byte offset
    return (unsigned)(unsigned long long)p;
}

// ---------------------------------------------------------------------------
// inp0[b,t,h] = bf16((x[b,t]*ff_w[h] + ff_b[h]) * (t < len[b]))
// ---------------------------------------------------------------------------
__global__ void embed_kernel(const float* __restrict__ x,
                             const int* __restrict__ lengths,
                             const float* __restrict__ ffw,
                             const float* __restrict__ ffb,
                             unsigned short* __restrict__ out) {
    int bt = blockIdx.x;
    int h  = threadIdx.x;
    int b  = bt >> 11;            // T_ = 2048
    int t  = bt & (T_ - 1);
    float v = 0.0f;
    if (t < lengths[b]) v = x[bt] * ffw[h] + ffb[h];
    out[(size_t)bt * H_ + h] = f2bf(v);
}

// ---------------------------------------------------------------------------
// Pack W = [w_ih | w_hh] (shape [512, K], K = din+128) fp32 -> bf16 in the
// WMMA B-matrix fragment layout:
//   idx = ((nt*ksteps + ks)*32 + lane)*16 + e
//   n = nt*16 + lane%16 ; k = ks*32 + (lane/16)*16 + e
// ---------------------------------------------------------------------------
__global__ void pack_w_kernel(const float* __restrict__ wih,
                              const float* __restrict__ whh,
                              int din, unsigned short* __restrict__ wp) {
    int K = din + H_;
    int ksteps = K >> 5;
    int idx = blockIdx.x * 256 + threadIdx.x;
    if (idx >= 512 * K) return;
    int e    = idx & 15;
    int lane = (idx >> 4) & 31;
    int rest = idx >> 9;
    int ks   = rest % ksteps;
    int nt   = rest / ksteps;
    int n = nt * 16 + (lane & 15);
    int k = ks * 32 + ((lane >> 4) << 4) + e;
    float v = (k < din) ? wih[(size_t)n * din + k]
                        : whh[(size_t)n * H_ + (k - din)];
    wp[idx] = f2bf(v);
}

__global__ void bias_sum_kernel(const float* __restrict__ bih,
                                const float* __restrict__ bhh,
                                float* __restrict__ bs) {
    int i = blockIdx.x * 256 + threadIdx.x;
    if (i < 512) bs[i] = bih[i] + bhh[i];
}

// ---------------------------------------------------------------------------
// Persistent LSTM scan. grid = 8 (4 M-tiles of 16 batch rows x 2 directions),
// block = 256 (8 waves). Wave w owns gate tiles nt = {w, w+8, w+16, w+24}
// (= gates i,f,g,o for hidden units [16w, 16w+16)) so the cell update is
// lane-local on the 16x16 fp32 C layout. c-state and ALL weight fragments
// stay in VGPRs (1024-VGPR MSB addressing); x tiles are double-buffered in
// LDS and filled with GLOBAL_LOAD_ASYNC_TO_LDS_B128 overlapped with WMMA.
// ---------------------------------------------------------------------------
template <int DIN>
__launch_bounds__(256, 1)
__global__ void lstm_scan_kernel(const unsigned short* __restrict__ in,
                                 const unsigned short* __restrict__ wp_f,
                                 const unsigned short* __restrict__ wp_b,
                                 const float* __restrict__ bs_f,
                                 const float* __restrict__ bs_b,
                                 const int* __restrict__ lengths,
                                 unsigned short* __restrict__ out) {
    constexpr int K       = DIN + H_;
    constexpr int KSTEPS  = K >> 5;
    constexpr int KS_X    = DIN >> 5;          // k-steps sourced from x tile
    constexpr int XSTRIDE = DIN + 8;           // pad: stride ≡ 4 dwords mod 64 banks
    constexpr int HSTRIDE = H_ + 8;
    constexpr int CHUNKS  = (16 * DIN) >> 3;   // 8-bf16 (16B) chunks per tile
    constexpr int REPS    = CHUNKS / 256;      // 1 (DIN=128) or 2 (DIN=256)

    __shared__ unsigned short s_x[2 * 16 * XSTRIDE];
    __shared__ unsigned short s_h[16 * HSTRIDE];
    __shared__ int s_len[16];

    const int tid   = threadIdx.x;
    const int lane  = tid & 31;
    const int w     = tid >> 5;
    const int mtile = blockIdx.x & 3;
    const int dir   = blockIdx.x >> 2;
    const int base  = mtile * 16;

    const unsigned short* wp = dir ? wp_b : wp_f;
    const float*          bs = dir ? bs_b : bs_f;

    if (tid < 16) s_len[tid] = lengths[base + tid];
    for (int i = tid; i < 16 * H_; i += 256)
        s_h[(i >> 7) * HSTRIDE + (i & 127)] = 0;

    const int u    = (w << 4) + (lane & 15);   // hidden unit owned by this lane
    const int lhi8 = (lane >> 4) << 3;         // 0 or 8 (C M-offset; A K-offset is lhi8*2)
    const int arow = lane & 15;

    const float bias_i = bs[u];
    const float bias_f = bs[128 + u];
    const float bias_g = bs[256 + u];
    const float bias_o = bs[384 + u];

    float cst[8];
#pragma unroll
    for (int v = 0; v < 8; ++v) cst[v] = 0.0f;

    // ---- hoist ALL weight fragments into registers (loop-invariant) ----
    Frag bfr[KSTEPS][4];
#pragma unroll
    for (int ks = 0; ks < KSTEPS; ++ks) {
#pragma unroll
        for (int g = 0; g < 4; ++g) {
            const unsigned short* p =
                wp + ((size_t)(w + 8 * g) * KSTEPS + ks) * 512 + lane * 16;
            bfr[ks][g].q[0] = *(const uint4*)p;
            bfr[ks][g].q[1] = *(const uint4*)(p + 8);
        }
    }

    __syncthreads();   // s_len / s_h ready

    // async-stage tile for timestep tn into buffer b_sel
    auto issue_tile = [&](int tn, int b_sel) {
#pragma unroll
        for (int rep = 0; rep < REPS; ++rep) {
            int i  = tid + rep * 256;
            int m  = i / (DIN >> 3);
            int k8 = (i - m * (DIN >> 3)) << 3;
            int len = s_len[m];
            unsigned short* dst = s_x + (b_sel * 16 + m) * XSTRIDE + k8;
            if (tn < len) {
                int tsrc = dir ? (len - 1 - tn) : tn;
                const unsigned short* src =
                    in + ((size_t)(base + m) * T_ + tsrc) * DIN + k8;
                unsigned ld = lds_addr32(dst);
                asm volatile("global_load_async_to_lds_b128 %0, %1, off"
                             :: "v"(ld), "v"((unsigned long long)src)
                             : "memory");
            } else {
                uint4 z = {0u, 0u, 0u, 0u};
                *(uint4*)dst = z;
            }
        }
    };

    issue_tile(0, 0);
    int buf = 0;

    for (int t = 0; t < T_; ++t) {
        asm volatile("s_wait_asynccnt 0x0" ::: "memory");  // tile[buf] landed
        __syncthreads();   // all tiles + h(t-1) visible to all waves

        if (t + 1 < T_) issue_tile(t + 1, buf ^ 1);        // overlap with WMMA

        // ---- gates = [x_t | h] @ W^T  via WMMA bf16, fp32 accumulate ----
        f32x8 acc0 = {}, acc1 = {}, acc2 = {}, acc3 = {};
#pragma unroll
        for (int ks = 0; ks < KSTEPS; ++ks) {
            Frag a;
            const unsigned short* ap;
            if (ks < KS_X)
                ap = s_x + (buf * 16 + arow) * XSTRIDE + (ks << 5) + (lhi8 << 1);
            else
                ap = s_h + arow * HSTRIDE + ((ks - KS_X) << 5) + (lhi8 << 1);
            a.q[0] = *(const uint4*)(ap);
            a.q[1] = *(const uint4*)(ap + 8);

            acc0 = __builtin_amdgcn_wmma_f32_16x16x32_bf16(false, a.v, false, bfr[ks][0].v, (short)0, acc0, false, false);
            acc1 = __builtin_amdgcn_wmma_f32_16x16x32_bf16(false, a.v, false, bfr[ks][1].v, (short)0, acc1, false, false);
            acc2 = __builtin_amdgcn_wmma_f32_16x16x32_bf16(false, a.v, false, bfr[ks][2].v, (short)0, acc2, false, false);
            acc3 = __builtin_amdgcn_wmma_f32_16x16x32_bf16(false, a.v, false, bfr[ks][3].v, (short)0, acc3, false, false);
        }
        __syncthreads();   // all A reads done before h is rewritten

        // ---- LSTM cell (lane-local), write h to LDS + layer output ----
#pragma unroll
        for (int v = 0; v < 8; ++v) {
            float gi = acc0[v] + bias_i;
            float gf = acc1[v] + bias_f;
            float gg = acc2[v] + bias_g;
            float go = acc3[v] + bias_o;
            float c  = sigmoidf_(gf) * cst[v] + sigmoidf_(gi) * tanh_fast(gg);
            cst[v]   = c;
            float h  = sigmoidf_(go) * tanh_fast(c);
            unsigned short hb = f2bf(h);
            int m = lhi8 + v;
            s_h[m * HSTRIDE + u] = hb;
            int len = s_len[m];
            int tst = dir ? ((t < len) ? (len - 1 - t) : t) : t;
            out[((size_t)(base + m) * T_ + tst) * (2 * H_) + dir * H_ + u] = hb;
        }
        buf ^= 1;
    }
}

// ---------------------------------------------------------------------------
// Masked mean / max / last pooling over rnn = out1 [B, T, 256].
// feat layout per batch row (864): [0,96) cavg | [96,352) mean |
// [352,608) max | [608,864) last.
// ---------------------------------------------------------------------------
__global__ void pool_kernel(const unsigned short* __restrict__ out1,
                            const int* __restrict__ lengths,
                            float* __restrict__ feat) {
    int b  = blockIdx.x;
    int ch = threadIdx.x;   // 256
    int len = lengths[b];
    const unsigned short* p = out1 + (size_t)b * T_ * 256 + ch;
    float s = 0.0f, mx = -3.4e38f;
    for (int t = 0; t < len; ++t) {
        float v = bf2f(p[(size_t)t * 256]);
        s += v;
        mx = fmaxf(mx, v);
    }
    float last = bf2f(p[(size_t)(len - 1) * 256]);
    float* fb = feat + b * 864;
    fb[96 + ch]  = s / (float)len;
    fb[352 + ch] = mx;
    fb[608 + ch] = last;
}

// ---------------------------------------------------------------------------
// 4-stage conv (1->4->8->16->96, k=3, VALID, ReLU) + time mean, chunked in
// LDS, deterministic per-thread accumulation (no atomics).
// ---------------------------------------------------------------------------
__global__ void convstack_kernel(const float* __restrict__ x,
                                 const float* __restrict__ cw0, const float* __restrict__ cb0,
                                 const float* __restrict__ cw1, const float* __restrict__ cb1,
                                 const float* __restrict__ cw2, const float* __restrict__ cb2,
                                 const float* __restrict__ cw3, const float* __restrict__ cb3,
                                 float* __restrict__ feat) {
    __shared__ float sx[264];
    __shared__ float sc1[4 * 262];
    __shared__ float sc2[8 * 260];
    __shared__ float sc3[16 * 258];
    const int b = blockIdx.x, tid = threadIdx.x;
    const int LOUT = T_ - 8;  // 2040
    float acc = 0.0f;

    for (int t0 = 0; t0 < LOUT; t0 += 256) {
        __syncthreads();
        for (int i = tid; i < 264; i += 256) {
            int t = t0 + i;
            sx[i] = (t < T_) ? x[(size_t)b * T_ + t] : 0.0f;
        }
        __syncthreads();
        for (int i = tid; i < 4 * 262; i += 256) {
            int c = i / 262, p = i % 262;
            float s = cb0[c];
#pragma unroll
            for (int k = 0; k < 3; ++k) s += cw0[c * 3 + k] * sx[p + k];
            sc1[c * 262 + p] = fmaxf(s, 0.0f);
        }
        __syncthreads();
        for (int i = tid; i < 8 * 260; i += 256) {
            int c = i / 260, p = i % 260;
            float s = cb1[c];
            for (int ic = 0; ic < 4; ++ic)
#pragma unroll
                for (int k = 0; k < 3; ++k)
                    s += cw1[(c * 4 + ic) * 3 + k] * sc1[ic * 262 + p + k];
            sc2[c * 260 + p] = fmaxf(s, 0.0f);
        }
        __syncthreads();
        for (int i = tid; i < 16 * 258; i += 256) {
            int c = i / 258, p = i % 258;
            float s = cb2[c];
            for (int ic = 0; ic < 8; ++ic)
#pragma unroll
                for (int k = 0; k < 3; ++k)
                    s += cw2[(c * 8 + ic) * 3 + k] * sc2[ic * 260 + p + k];
            sc3[c * 258 + p] = fmaxf(s, 0.0f);
        }
        __syncthreads();
        if (tid < 96) {
            for (int p = 0; p < 256; ++p) {
                int t = t0 + p;
                if (t < LOUT) {
                    float s = cb3[tid];
                    for (int ic = 0; ic < 16; ++ic)
#pragma unroll
                        for (int k = 0; k < 3; ++k)
                            s += cw3[(tid * 16 + ic) * 3 + k] * sc3[ic * 258 + p + k];
                    acc += fmaxf(s, 0.0f);
                }
            }
        }
    }
    if (tid < 96) feat[b * 864 + tid] = acc / (float)LOUT;
}

// ---------------------------------------------------------------------------
// out[b,n] = lin_b[n] + lin_w[n,:] . feat[b,:]   (64x4 = 256 threads)
// ---------------------------------------------------------------------------
__global__ void final_linear_kernel(const float* __restrict__ feat,
                                    const float* __restrict__ lw,
                                    const float* __restrict__ lb,
                                    float* __restrict__ out) {
    int i = threadIdx.x;
    int b = i >> 2, n = i & 3;
    float s = lb[n];
    const float* f = feat + b * 864;
    const float* wr = lw + n * 864;
    for (int k = 0; k < 864; ++k) s += wr[k] * f[k];
    out[b * 4 + n] = s;
}

// ---------------------------------------------------------------------------
extern "C" void kernel_launch(void* const* d_in, const int* in_sizes, int n_in,
                              void* d_out, int out_size, void* d_ws, size_t ws_size,
                              hipStream_t stream) {
    (void)in_sizes; (void)n_in; (void)out_size; (void)ws_size;

    const float* x       = (const float*)d_in[0];
    const int*   lengths = (const int*)d_in[1];
    const float* ff_w    = (const float*)d_in[2];
    const float* ff_b    = (const float*)d_in[3];
    const float* w_ih_l0f = (const float*)d_in[4];
    const float* w_hh_l0f = (const float*)d_in[5];
    const float* b_ih_l0f = (const float*)d_in[6];
    const float* b_hh_l0f = (const float*)d_in[7];
    const float* w_ih_l0b = (const float*)d_in[8];
    const float* w_hh_l0b = (const float*)d_in[9];
    const float* b_ih_l0b = (const float*)d_in[10];
    const float* b_hh_l0b = (const float*)d_in[11];
    const float* w_ih_l1f = (const float*)d_in[12];
    const float* w_hh_l1f = (const float*)d_in[13];
    const float* b_ih_l1f = (const float*)d_in[14];
    const float* b_hh_l1f = (const float*)d_in[15];
    const float* w_ih_l1b = (const float*)d_in[16];
    const float* w_hh_l1b = (const float*)d_in[17];
    const float* b_ih_l1b = (const float*)d_in[18];
    const float* b_hh_l1b = (const float*)d_in[19];
    const float* cw0 = (const float*)d_in[20];
    const float* cb0 = (const float*)d_in[21];
    const float* cw1 = (const float*)d_in[22];
    const float* cb1 = (const float*)d_in[23];
    const float* cw2 = (const float*)d_in[24];
    const float* cb2 = (const float*)d_in[25];
    const float* cw3 = (const float*)d_in[26];
    const float* cb3 = (const float*)d_in[27];
    const float* lin_w = (const float*)d_in[28];
    const float* lin_b = (const float*)d_in[29];

    unsigned char* ws = (unsigned char*)d_ws;
    size_t off = 0;
    auto take = [&](size_t bytes) -> void* {
        void* p = ws + off;
        off = (off + bytes + 255) & ~(size_t)255;
        return p;
    };

    unsigned short* inp0 = (unsigned short*)take((size_t)B_ * T_ * H_ * 2);
    unsigned short* out0 = (unsigned short*)take((size_t)B_ * T_ * 256 * 2);
    unsigned short* out1 = (unsigned short*)take((size_t)B_ * T_ * 256 * 2);
    unsigned short* wp_l0f = (unsigned short*)take((size_t)512 * 256 * 2);
    unsigned short* wp_l0b = (unsigned short*)take((size_t)512 * 256 * 2);
    unsigned short* wp_l1f = (unsigned short*)take((size_t)512 * 384 * 2);
    unsigned short* wp_l1b = (unsigned short*)take((size_t)512 * 384 * 2);
    float* bs_l0f = (float*)take(512 * 4);
    float* bs_l0b = (float*)take(512 * 4);
    float* bs_l1f = (float*)take(512 * 4);
    float* bs_l1b = (float*)take(512 * 4);
    float* feat   = (float*)take((size_t)B_ * 864 * 4);

    embed_kernel<<<dim3(B_ * T_), dim3(H_), 0, stream>>>(x, lengths, ff_w, ff_b, inp0);

    pack_w_kernel<<<dim3(512), dim3(256), 0, stream>>>(w_ih_l0f, w_hh_l0f, 128, wp_l0f);
    pack_w_kernel<<<dim3(512), dim3(256), 0, stream>>>(w_ih_l0b, w_hh_l0b, 128, wp_l0b);
    pack_w_kernel<<<dim3(768), dim3(256), 0, stream>>>(w_ih_l1f, w_hh_l1f, 256, wp_l1f);
    pack_w_kernel<<<dim3(768), dim3(256), 0, stream>>>(w_ih_l1b, w_hh_l1b, 256, wp_l1b);

    bias_sum_kernel<<<dim3(2), dim3(256), 0, stream>>>(b_ih_l0f, b_hh_l0f, bs_l0f);
    bias_sum_kernel<<<dim3(2), dim3(256), 0, stream>>>(b_ih_l0b, b_hh_l0b, bs_l0b);
    bias_sum_kernel<<<dim3(2), dim3(256), 0, stream>>>(b_ih_l1f, b_hh_l1f, bs_l1f);
    bias_sum_kernel<<<dim3(2), dim3(256), 0, stream>>>(b_ih_l1b, b_hh_l1b, bs_l1b);

    lstm_scan_kernel<128><<<dim3(8), dim3(256), 0, stream>>>(
        inp0, wp_l0f, wp_l0b, bs_l0f, bs_l0b, lengths, out0);
    lstm_scan_kernel<256><<<dim3(8), dim3(256), 0, stream>>>(
        out0, wp_l1f, wp_l1b, bs_l1f, bs_l1b, lengths, out1);

    pool_kernel<<<dim3(B_), dim3(256), 0, stream>>>(out1, lengths, feat);
    convstack_kernel<<<dim3(B_), dim3(256), 0, stream>>>(
        x, cw0, cb0, cw1, cb1, cw2, cb2, cw3, cb3, feat);
    final_linear_kernel<<<dim3(1), dim3(256), 0, stream>>>(feat, lin_w, lin_b, (float*)d_out);
}
